// Qwen3Attention_87187836109410
// MI455X (gfx1250) — matrix operand order repaired and checked by reading, not compiled, not run on hardware
//
#include <hip/hip_runtime.h>
#include <hip/hip_bf16.h>

// ---------------------------------------------------------------------------
// Qwen3 attention block for MI455X (gfx1250, wave32, WMMA bf16, async-LDS)
//   B=1, S=2048, HID=4096, H=32, KV=8, D=128, GROUPS=4
// ---------------------------------------------------------------------------

typedef __bf16 bf16_t;
typedef __attribute__((ext_vector_type(8)))  __bf16 v8bf;
typedef __attribute__((ext_vector_type(16))) __bf16 v16bf;
typedef __attribute__((ext_vector_type(8)))  float  v8f;

constexpr int   SEQ   = 2048;
constexpr int   HIDN  = 4096;
constexpr int   NHQ   = 32;
constexpr int   NHKV  = 8;
constexpr int   HD    = 128;
constexpr float EPSV  = 1e-6f;
constexpr float SCL   = 0.08838834764831845f; // 128^-0.5

__device__ __forceinline__ v16bf cat16(v8bf lo, v8bf hi) {
    return __builtin_shufflevector(lo, hi, 0,1,2,3,4,5,6,7,8,9,10,11,12,13,14,15);
}

__device__ __forceinline__ v8f splat8(float x) {
    v8f r;
#pragma unroll
    for (int i = 0; i < 8; ++i) r[i] = x;
    return r;
}

__device__ __forceinline__ v8f wmma_bf16(v16bf a, v16bf b, v8f c) {
    // D(f32 16x16) = A(bf16 16x32) * B(bf16 32x16) + C
    return __builtin_amdgcn_wmma_f32_16x16x32_bf16(
        /*neg_a=*/false, a, /*neg_b=*/false, b,
        /*c_mod=*/(short)0, c, /*reuse_a=*/false, /*reuse_b=*/false);
}

// Wave-internal LDS ordering fence (DS pipe is in-order per wave).
__device__ __forceinline__ void lds_fence() {
    __builtin_amdgcn_wave_barrier();
    asm volatile("s_wait_dscnt 0" ::: "memory");
}

// Async global->LDS 16B copy (ASYNCcnt-tracked, CDNA5). LDS byte address is
// the low 32 bits of the flat shared pointer (LDS aperture truncation).
__device__ __forceinline__ void async_copy_b128(void* lds_ptr, const void* gptr) {
    unsigned int       ldsa = (unsigned int)(uintptr_t)lds_ptr;
    unsigned long long ga   = (unsigned long long)(uintptr_t)gptr;
    asm volatile("global_load_async_to_lds_b128 %0, %1, off"
                 :: "v"(ldsa), "v"(ga) : "memory");
}

__device__ __forceinline__ void wait_async0() {
    asm volatile("s_wait_asynccnt 0" ::: "memory");
}

// ---------------------------------------------------------------------------
// fp32 -> bf16 elementwise convert
// ---------------------------------------------------------------------------
__global__ void f32_to_bf16(const float* __restrict__ src,
                            bf16_t* __restrict__ dst, int n) {
    int i = blockIdx.x * blockDim.x + threadIdx.x;
    int stride = gridDim.x * blockDim.x;
    for (; i < n; i += stride) dst[i] = (bf16_t)src[i];
}

// ---------------------------------------------------------------------------
// C[M,N](f32) = A[M,K](bf16, row-major) * W[N,K](bf16, row-major)^T
// Block = 128 threads (4 waves) -> 128x64 output tile.
//   - Shared 64x32 B tile staged in LDS via async global->LDS, double-buffered
//     (B fetched once per block instead of once per wave).
//   - Each wave computes a 32x64 subtile: 8 WMMAs per 32-K step.
//   - All 4 B-fragments pre-loaded into distinct registers so the DS loads
//     pipeline ahead of the back-to-back WMMA burst.
// grid = (N/64, M/128), block = 128
// ---------------------------------------------------------------------------
__global__ __launch_bounds__(128)
void gemm_bf16_wmma(const bf16_t* __restrict__ A,
                    const bf16_t* __restrict__ W,
                    float* __restrict__ C,
                    int M, int N, int K) {
    const int tid  = threadIdx.x;
    const int wave = tid >> 5;
    const int lane = tid & 31;
    const int half = lane >> 4;     // K-subchunk selector
    const int mrow = lane & 15;     // row (A) / col (B,C)
    const int tm = blockIdx.y * 128 + wave * 32;
    const int tn = blockIdx.x * 64;

    __shared__ bf16_t Bs[2][64][32];   // [buf][n-row][k] , 4KB per buffer

    // Stage one 64x32 B tile: 256 16B chunks, 2 per thread, via async DMA.
    auto stage = [&](int buf, int k0) {
        {
            const int c = tid;           // chunk 0..127
            const int r = c >> 2, kk = (c & 3) * 8;
            async_copy_b128(&Bs[buf][r][kk], W + (size_t)(tn + r) * K + k0 + kk);
        }
        {
            const int c = tid + 128;     // chunk 128..255
            const int r = c >> 2, kk = (c & 3) * 8;
            async_copy_b128(&Bs[buf][r][kk], W + (size_t)(tn + r) * K + k0 + kk);
        }
    };

    v8f acc0[4], acc1[4];
#pragma unroll
    for (int t = 0; t < 4; ++t) { acc0[t] = splat8(0.0f); acc1[t] = splat8(0.0f); }

    stage(0, 0);
    int buf = 0;
    for (int k0 = 0; k0 < K; k0 += 32) {
        wait_async0();        // current buffer landed in LDS
        __syncthreads();      // ...for every wave (also: prev compute done)
        if (k0 + 32 < K) stage(buf ^ 1, k0 + 32);   // overlap next DMA

        // A fragments (16x32 bf16): lane<16 holds K {0..7,16..23}, lane>=16 {8..15,24..31}
        const bf16_t* ap0 = A + (size_t)(tm + mrow) * K + k0 + half * 8;
        v16bf af0 = cat16(*(const v8bf*)ap0, *(const v8bf*)(ap0 + 16));
        const bf16_t* ap1 = ap0 + (size_t)16 * K;
        v16bf af1 = cat16(*(const v8bf*)ap1, *(const v8bf*)(ap1 + 16));

        // Pre-load all 4 B fragments (32x16 each): lane holds column
        // n = tn+t*16+mrow, K {half*8..+7, 16+half*8..+7} (two 16B runs in LDS).
        v16bf bfr[4];
#pragma unroll
        for (int t = 0; t < 4; ++t) {
            const bf16_t* bp = &Bs[buf][t * 16 + mrow][half * 8];
            bfr[t] = cat16(*(const v8bf*)bp, *(const v8bf*)(bp + 16));
        }
#pragma unroll
        for (int t = 0; t < 4; ++t) {
            acc0[t] = wmma_bf16(af0, bfr[t], acc0[t]);
            acc1[t] = wmma_bf16(af1, bfr[t], acc1[t]);
        }
        buf ^= 1;
    }

#pragma unroll
    for (int t = 0; t < 4; ++t)
#pragma unroll
        for (int v = 0; v < 8; ++v) {
            C[(size_t)(tm + v + half * 8) * N + tn + t * 16 + mrow]      = acc0[t][v];
            C[(size_t)(tm + 16 + v + half * 8) * N + tn + t * 16 + mrow] = acc1[t][v];
        }
}

// ---------------------------------------------------------------------------
// Per-head RMSNorm (over D=128) + RoPE, fp32 in -> bf16 out
// grid = (S, NH), block = 128 (one thread per d)
// ---------------------------------------------------------------------------
__global__ __launch_bounds__(128)
void qk_norm_rope(const float* __restrict__ X,   // [S, NH*D]
                  const float* __restrict__ w,   // [D]
                  const float* __restrict__ cosb,// [S, D]
                  const float* __restrict__ sinb,// [S, D]
                  bf16_t* __restrict__ out,      // [S, NH*D]
                  int NH) {
    const int s = blockIdx.x;
    const int h = blockIdx.y;
    const int d = threadIdx.x;

    __shared__ float red[128];
    __shared__ float sh[128];

    const size_t base = ((size_t)s * NH + h) * HD;
    float x = X[base + d];

    red[d] = x * x;
    __syncthreads();
#pragma unroll
    for (int off = 64; off > 0; off >>= 1) {
        if (d < off) red[d] += red[d + off];
        __syncthreads();
    }
    const float mean = red[0] * (1.0f / (float)HD);
    const float r = rsqrtf(mean + EPSV);
    const float xn = x * r * w[d];

    sh[d] = xn;
    __syncthreads();
    const float rot = (d < 64) ? -sh[d + 64] : sh[d - 64];

    const float c = cosb[(size_t)s * HD + d];
    const float sn = sinb[(size_t)s * HD + d];
    out[base + d] = (bf16_t)(xn * c + rot * sn);
}

// ---------------------------------------------------------------------------
// Causal GQA flash attention (bf16 operands, fp32 online softmax)
// One wave per (16-query tile, head). Key chunks of 32.
//   QK^T : 8 WMMAs/chunk (Q frags cached, K frags straight from global)
//   PV   : 8 WMMAs/chunk (P via LDS re-layout, V staged transposed in LDS)
// grid = (S/16, H), block = 32
// ---------------------------------------------------------------------------
__global__ __launch_bounds__(32)
void attn_fwd(const bf16_t* __restrict__ Qb,  // [S, H*D] bf16 (post norm+rope)
              const bf16_t* __restrict__ Kb,  // [S, KV*D]
              const bf16_t* __restrict__ Vb,  // [S, KV*D]
              bf16_t* __restrict__ Ob) {      // [S, H*D] bf16
    const int lane = threadIdx.x & 31;
    const int half = lane >> 4;
    const int mrow = lane & 15;
    const int qbase = blockIdx.x * 16;
    const int h = blockIdx.y;
    const int kvh = h >> 2;                    // GROUPS = 4

    __shared__ bf16_t Pt[16][32];              // P tile, row-major [q][key]
    __shared__ bf16_t Vt[HD][32];              // V^T tile [d][key]

    // Cache Q A-fragments for all 4 K-steps over D=128
    v16bf qf[4];
#pragma unroll
    for (int ks = 0; ks < 4; ++ks) {
        const bf16_t* qp = Qb + (size_t)(qbase + mrow) * (NHQ * HD) + h * HD + ks * 32 + half * 8;
        qf[ks] = cat16(*(const v8bf*)qp, *(const v8bf*)(qp + 16));
    }

    v8f acc[8];
#pragma unroll
    for (int t = 0; t < 8; ++t) acc[t] = splat8(0.0f);
    v8f m_run = splat8(-3.0e38f);
    v8f l_run = splat8(0.0f);

    const int nchunks = (qbase + 47) >> 5;     // keys 0 .. qbase+15 covered
    for (int c = 0; c < nchunks; ++c) {
        const int kb = c * 32;

        // ---- scores: 16 queries x 32 keys --------------------------------
        v8f s0 = splat8(0.0f), s1 = splat8(0.0f);
#pragma unroll
        for (int ks = 0; ks < 4; ++ks) {
            const bf16_t* kp0 = Kb + (size_t)(kb + mrow) * (NHKV * HD) + kvh * HD + ks * 32 + half * 8;
            v16bf kf0 = cat16(*(const v8bf*)kp0, *(const v8bf*)(kp0 + 16));
            s0 = wmma_bf16(qf[ks], kf0, s0);
            const bf16_t* kp1 = kp0 + (size_t)16 * (NHKV * HD);
            v16bf kf1 = cat16(*(const v8bf*)kp1, *(const v8bf*)(kp1 + 16));
            s1 = wmma_bf16(qf[ks], kf1, s1);
        }

        // ---- stage V^T for this chunk (lane = key) -----------------------
        {
            const bf16_t* vp = Vb + (size_t)(kb + lane) * (NHKV * HD) + kvh * HD;
#pragma unroll
            for (int d0 = 0; d0 < HD; d0 += 8) {
                v8bf vv = *(const v8bf*)(vp + d0);
#pragma unroll
                for (int j = 0; j < 8; ++j) Vt[d0 + j][lane] = vv[j];
            }
        }

        // ---- mask + online softmax (fp32) --------------------------------
        v8f p0, p1, alpha;
#pragma unroll
        for (int v = 0; v < 8; ++v) {
            const int qrow = qbase + v + half * 8;
            float x0 = s0[v] * SCL; if (kb + mrow > qrow)      x0 = -3.0e38f;
            float x1 = s1[v] * SCL; if (kb + 16 + mrow > qrow) x1 = -3.0e38f;
            // row max across the 16 lanes of this half (stays within half)
            float t = fmaxf(x0, x1);
            t = fmaxf(t, __shfl_xor(t, 1, 32));
            t = fmaxf(t, __shfl_xor(t, 2, 32));
            t = fmaxf(t, __shfl_xor(t, 4, 32));
            t = fmaxf(t, __shfl_xor(t, 8, 32));
            const float mn = fmaxf(m_run[v], t);
            const float a  = __expf(m_run[v] - mn);
            const float e0 = __expf(x0 - mn);
            const float e1 = __expf(x1 - mn);
            float rsum = e0 + e1;
            rsum += __shfl_xor(rsum, 1, 32);
            rsum += __shfl_xor(rsum, 2, 32);
            rsum += __shfl_xor(rsum, 4, 32);
            rsum += __shfl_xor(rsum, 8, 32);
            l_run[v] = l_run[v] * a + rsum;
            m_run[v] = mn;
            alpha[v] = a;
            p0[v] = e0;
            p1[v] = e1;
        }
#pragma unroll
        for (int t = 0; t < 8; ++t)
#pragma unroll
            for (int v = 0; v < 8; ++v) acc[t][v] *= alpha[v];

        // ---- P: C-frag layout -> LDS -> A-frag layout --------------------
#pragma unroll
        for (int v = 0; v < 8; ++v) {
            Pt[v + half * 8][mrow]      = (bf16_t)p0[v];
            Pt[v + half * 8][16 + mrow] = (bf16_t)p1[v];
        }
        lds_fence();
        const bf16_t* pr = &Pt[mrow][0];
        v16bf pa = cat16(*(const v8bf*)(pr + half * 8),
                         *(const v8bf*)(pr + 16 + half * 8));

        // ---- PV: 8 d-subtiles in two pre-loaded groups of 4 --------------
#pragma unroll
        for (int g = 0; g < 2; ++g) {
            v16bf vfr[4];
#pragma unroll
            for (int t = 0; t < 4; ++t) {
                const bf16_t* vr = &Vt[(g * 4 + t) * 16 + mrow][0] + half * 8;
                vfr[t] = cat16(*(const v8bf*)vr, *(const v8bf*)(vr + 16));
            }
#pragma unroll
            for (int t = 0; t < 4; ++t)
                acc[g * 4 + t] = wmma_bf16(pa, vfr[t], acc[g * 4 + t]);
        }
        lds_fence();   // keep this chunk's LDS reads ahead of next-chunk writes
    }

    // ---- normalize and emit bf16 attn output ([S, H*D]) -------------------
#pragma unroll
    for (int t = 0; t < 8; ++t)
#pragma unroll
        for (int v = 0; v < 8; ++v) {
            const float val = acc[t][v] / l_run[v];
            Ob[(size_t)(qbase + v + half * 8) * (NHQ * HD) + h * HD + t * 16 + mrow] = (bf16_t)val;
        }
}

// ---------------------------------------------------------------------------
// Host-side orchestration
// ---------------------------------------------------------------------------
extern "C" void kernel_launch(void* const* d_in, const int* in_sizes, int n_in,
                              void* d_out, int out_size, void* d_ws, size_t ws_size,
                              hipStream_t stream) {
    (void)in_sizes; (void)n_in; (void)out_size; (void)ws_size;

    const float* hs   = (const float*)d_in[0];
    const float* cosb = (const float*)d_in[1];
    const float* sinb = (const float*)d_in[2];
    const float* Wq   = (const float*)d_in[3];
    const float* Wk   = (const float*)d_in[4];
    const float* Wv   = (const float*)d_in[5];
    const float* Wo   = (const float*)d_in[6];
    const float* qnw  = (const float*)d_in[7];
    const float* knw  = (const float*)d_in[8];
    float* out = (float*)d_out;

    char* ws = (char*)d_ws;
    size_t off = 0;
    auto alloc = [&](size_t bytes) { char* p = ws + off; off += (bytes + 255) & ~(size_t)255; return p; };

    bf16_t* hsb  = (bf16_t*)alloc((size_t)SEQ * HIDN * 2);          // hidden bf16
    bf16_t* wqb  = (bf16_t*)alloc((size_t)NHQ * HD * HIDN * 2);     // Wq bf16
    bf16_t* wkb  = (bf16_t*)alloc((size_t)NHKV * HD * HIDN * 2);    // Wk bf16
    bf16_t* wvb  = (bf16_t*)alloc((size_t)NHKV * HD * HIDN * 2);    // Wv bf16
    bf16_t* wob  = (bf16_t*)alloc((size_t)HIDN * NHQ * HD * 2);     // Wo bf16
    float*  qf   = (float*) alloc((size_t)SEQ * NHQ * HD * 4);      // Q fp32
    float*  kf   = (float*) alloc((size_t)SEQ * NHKV * HD * 4);     // K fp32
    float*  vf   = (float*) alloc((size_t)SEQ * NHKV * HD * 4);     // V fp32
    bf16_t* qb2  = (bf16_t*)alloc((size_t)SEQ * NHQ * HD * 2);      // Q bf16 (norm+rope)
    bf16_t* kb2  = (bf16_t*)alloc((size_t)SEQ * NHKV * HD * 2);     // K bf16 (norm+rope)
    bf16_t* vb2  = (bf16_t*)alloc((size_t)SEQ * NHKV * HD * 2);     // V bf16
    bf16_t* ab   = (bf16_t*)alloc((size_t)SEQ * NHQ * HD * 2);      // attn out bf16

    // 1) fp32 -> bf16 conversions
    f32_to_bf16<<<2048, 256, 0, stream>>>(hs, hsb, SEQ * HIDN);
    f32_to_bf16<<<2048, 256, 0, stream>>>(Wq, wqb, NHQ * HD * HIDN);
    f32_to_bf16<<<1024, 256, 0, stream>>>(Wk, wkb, NHKV * HD * HIDN);
    f32_to_bf16<<<1024, 256, 0, stream>>>(Wv, wvb, NHKV * HD * HIDN);
    f32_to_bf16<<<2048, 256, 0, stream>>>(Wo, wob, HIDN * NHQ * HD);

    // 2) QKV projections (WMMA bf16, fp32 accum, async-LDS weight staging)
    gemm_bf16_wmma<<<dim3(NHQ * HD / 64,  SEQ / 128), 128, 0, stream>>>(hsb, wqb, qf, SEQ, NHQ * HD,  HIDN);
    gemm_bf16_wmma<<<dim3(NHKV * HD / 64, SEQ / 128), 128, 0, stream>>>(hsb, wkb, kf, SEQ, NHKV * HD, HIDN);
    gemm_bf16_wmma<<<dim3(NHKV * HD / 64, SEQ / 128), 128, 0, stream>>>(hsb, wvb, vf, SEQ, NHKV * HD, HIDN);

    // 3) per-head RMSNorm + RoPE -> bf16; V straight convert
    qk_norm_rope<<<dim3(SEQ, NHQ),  128, 0, stream>>>(qf, qnw, cosb, sinb, qb2, NHQ);
    qk_norm_rope<<<dim3(SEQ, NHKV), 128, 0, stream>>>(kf, knw, cosb, sinb, kb2, NHKV);
    f32_to_bf16<<<1024, 256, 0, stream>>>(vf, vb2, SEQ * NHKV * HD);

    // 4) causal GQA flash attention
    attn_fwd<<<dim3(SEQ / 16, NHQ), 32, 0, stream>>>(qb2, kb2, vb2, ab);

    // 5) output projection -> fp32 d_out
    gemm_bf16_wmma<<<dim3(HIDN / 64, SEQ / 128), 128, 0, stream>>>(ab, wob, out, SEQ, HIDN, NHQ * HD);
}
